// GAT2_89008902243173
// MI455X (gfx1250) — matrix operand order repaired and checked by reading, not compiled, not run on hardware
//
#include <hip/hip_runtime.h>
#include <hip/hip_bf16.h>
#include <math.h>

typedef __bf16 v16bf __attribute__((ext_vector_type(16)));
typedef __bf16 v8bf  __attribute__((ext_vector_type(8)));
typedef __bf16 v4bf  __attribute__((ext_vector_type(4)));
typedef float  v8f   __attribute__((ext_vector_type(8)));
typedef int    v4i   __attribute__((ext_vector_type(4)));

#define NEG_SLOPE 0.2f

#if defined(__AMDGCN__) && __has_builtin(__builtin_amdgcn_global_load_async_to_lds_b128)
#define USE_ASYNC_LDS 1
#else
#define USE_ASYNC_LDS 0
#endif

// ---------------------------------------------------------------------------
// Utility: fill a float buffer with a constant
// ---------------------------------------------------------------------------
__global__ void fill_f32(float* __restrict__ p, float v, int n) {
    int i = blockIdx.x * blockDim.x + threadIdx.x;
    if (i < n) p[i] = v;
}

// ---------------------------------------------------------------------------
// Elementwise f32 -> bf16 conversion (vectorized: float4 in, 8B out)
// ---------------------------------------------------------------------------
__global__ void cvt_f32_bf16(const float* __restrict__ in, __bf16* __restrict__ out, int n4) {
    int i = blockIdx.x * blockDim.x + threadIdx.x;
    if (i >= n4) return;
    const int base = i * 4;
    float4 v = *(const float4*)(in + base);
    v4bf o;
    o[0] = (__bf16)v.x; o[1] = (__bf16)v.y; o[2] = (__bf16)v.z; o[3] = (__bf16)v.w;
    *(v4bf*)(out + base) = o;
}

// ---------------------------------------------------------------------------
// Pre-pack W[KDIM x NDIM] (f32, row-major) into bf16 WMMA B-fragment order:
// Wp[((kt*NCT + ct)*32 + lane)*16 + e] = W[(kt*32 + (lane>>4)*16 + e)*NDIM + ct*16 + (lane&15)]
// so each lane's B fragment is 16 contiguous bf16 (32B, aligned).
// ---------------------------------------------------------------------------
template <int KDIM, int NDIM>
__global__ void prepack_w(const float* __restrict__ W, __bf16* __restrict__ Wp) {
    int i = blockIdx.x * blockDim.x + threadIdx.x;
    if (i >= KDIM * NDIM) return;
    constexpr int NCT = NDIM / 16;
    const int e    = i & 15;
    const int lane = (i >> 4) & 31;
    const int fct  = i >> 9;          // kt*NCT + ct
    const int ct   = fct % NCT;
    const int kt   = fct / NCT;
    const int k    = kt * 32 + (lane >> 4) * 16 + e;
    const int n    = ct * 16 + (lane & 15);
    Wp[i] = (__bf16)W[k * NDIM + n];
}

// ---------------------------------------------------------------------------
// Sign-aware float atomic max via integer atomics (monotone bit patterns)
// ---------------------------------------------------------------------------
__device__ __forceinline__ void atomicMaxF(float* addr, float v) {
    if (v >= 0.0f) {
        atomicMax((int*)addr, __float_as_int(v));
    } else {
        atomicMin((unsigned int*)addr, (unsigned int)__float_as_int(v));
    }
}

// ---------------------------------------------------------------------------
// WMMA GEMM: C[M x NDIM] = A[M x KDIM] * W[KDIM x NDIM]
// A: bf16 row-major (pre-converted). Wp: bf16 pre-packed fragment order.
// Block = 128 threads = 4 waves; each wave owns a 16-row strip across all
// NDIM/16 column tiles. Wp staged to LDS via async-to-LDS when available.
// ---------------------------------------------------------------------------
template <int KDIM, int NDIM>
__global__ void gemm_wmma_bf16(const __bf16* __restrict__ A,
                               const __bf16* __restrict__ Wp,
                               float* __restrict__ C, int M) {
    __shared__ __bf16 ldsW[KDIM * NDIM];
    const int tid = threadIdx.x;

#if USE_ASYNC_LDS
    {
        typedef __attribute__((address_space(1))) v4i* gp_t;
        typedef __attribute__((address_space(3))) v4i* lp_t;
        gp_t gsrc = (gp_t)Wp;
        lp_t ldst = (lp_t)ldsW;
        const int n16 = KDIM * NDIM * 2 / 16;
        for (int i = tid; i < n16; i += 128)
            __builtin_amdgcn_global_load_async_to_lds_b128(gsrc + i, ldst + i, 0, 0);
#if __has_builtin(__builtin_amdgcn_s_wait_asynccnt)
        __builtin_amdgcn_s_wait_asynccnt(0);
#else
        asm volatile("s_wait_asynccnt 0x0" ::: "memory");
#endif
    }
#else
    for (int i = tid * 8; i < KDIM * NDIM; i += 128 * 8)
        *(v8bf*)(ldsW + i) = *(const v8bf*)(Wp + i);
#endif
    __syncthreads();

    const int lane = tid & 31;
    const int wave = tid >> 5;
    const int r16  = lane & 15;
    const int hi   = lane >> 4;            // 0: lanes 0-15, 1: lanes 16-31
    const int rowBase = blockIdx.x * 64 + wave * 16;

    int row = rowBase + r16;
    if (row >= M) row = M - 1;             // clamp; stores guarded below
    const __bf16* arow = A + (size_t)row * KDIM;

    constexpr int NCT = NDIM / 16;
    v8f acc[NCT] = {};

#pragma unroll
    for (int kstep = 0; kstep < KDIM; kstep += 32) {
        // A fragment: two contiguous 8-elem bf16 runs -> two b128 loads
        const int kb = kstep + hi * 16;
        v8bf a0 = *(const v8bf*)(arow + kb);
        v8bf a1 = *(const v8bf*)(arow + kb + 8);
        v16bf af = __builtin_shufflevector(a0, a1, 0, 1, 2, 3, 4, 5, 6, 7,
                                           8, 9, 10, 11, 12, 13, 14, 15);
        const int kt = kstep >> 5;
#pragma unroll
        for (int ct = 0; ct < NCT; ++ct) {
            // B fragment: 32B contiguous per lane in packed LDS
            v16bf bf = *(const v16bf*)(ldsW + (((kt * NCT + ct) * 32 + lane) << 4));
            acc[ct] = __builtin_amdgcn_wmma_f32_16x16x32_bf16(
                false, af, false, bf, (short)0, acc[ct], false, false);
        }
    }

    // D layout: VGPR r -> (M = rowBase + r + hi*8, N = ct*16 + r16)
#pragma unroll
    for (int ct = 0; ct < NCT; ++ct) {
        const int n = ct * 16 + r16;
#pragma unroll
        for (int r = 0; r < 8; ++r) {
            const int m = rowBase + hi * 8 + r;
            if (m < M) C[(size_t)m * NDIM + n] = acc[ct][r];
        }
    }
}

// ---------------------------------------------------------------------------
// Per-node attention dots: as_[i] = h[i]·a_src ; ad_[i] = h[i]·a_dst
// ---------------------------------------------------------------------------
template <int F>
__global__ void attdot(const float* __restrict__ h,
                       const float* __restrict__ a_s,
                       const float* __restrict__ a_d,
                       float* __restrict__ as_, float* __restrict__ ad_, int n) {
    int i = blockIdx.x * blockDim.x + threadIdx.x;
    if (i >= n) return;
    const float* hr = h + (size_t)i * F;
    float s = 0.f, d = 0.f;
#pragma unroll
    for (int f = 0; f < F; ++f) {
        float v = hr[f];
        s = fmaf(v, a_s[f], s);
        d = fmaf(v, a_d[f], d);
    }
    as_[i] = s;
    ad_[i] = d;
}

// ---------------------------------------------------------------------------
// Edge pass 1: e = leaky_relu(as[src] + ad[dst]); segment max over dst
// ---------------------------------------------------------------------------
__global__ void edge_score_max(const int* __restrict__ ei, int E, int N,
                               const float* __restrict__ as_,
                               const float* __restrict__ ad_,
                               float* __restrict__ ebuf, float* __restrict__ maxv) {
    int e = blockIdx.x * blockDim.x + threadIdx.x;
    const int ET = E + N;
    if (e >= ET) return;
    int s, d;
    if (e < E) { s = ei[e]; d = ei[E + e]; } else { s = d = e - E; }
    float v = as_[s] + ad_[d];
    v = (v >= 0.f) ? v : NEG_SLOPE * v;
    ebuf[e] = v;
    atomicMaxF(maxv + d, v);
}

// ---------------------------------------------------------------------------
// Edge pass 2: p = exp(e - m[dst]); segment sum over dst; ebuf <- p
// ---------------------------------------------------------------------------
__global__ void edge_exp_sum(const int* __restrict__ ei, int E, int N,
                             const float* __restrict__ maxv,
                             float* __restrict__ ebuf, float* __restrict__ denom) {
    int e = blockIdx.x * blockDim.x + threadIdx.x;
    const int ET = E + N;
    if (e >= ET) return;
    int d;
    if (e < E) { d = ei[E + e]; } else { d = e - E; }
    float p = __expf(ebuf[e] - maxv[d]);
    ebuf[e] = p;
    atomicAdd(denom + d, p);
}

// ---------------------------------------------------------------------------
// Edge pass 3: agg[dst] += (p/denom[dst]) * h[src]; one wave per edge,
// lanes stride across F features (coalesced), float atomics into L2.
// ---------------------------------------------------------------------------
template <int F>
__global__ void edge_aggregate(const int* __restrict__ ei, int E, int N,
                               const float* __restrict__ h,
                               const float* __restrict__ ebuf,
                               const float* __restrict__ denom,
                               float* __restrict__ agg) {
    int gt = blockIdx.x * blockDim.x + threadIdx.x;
    int e = gt >> 5;
    int lane = gt & 31;
    const int ET = E + N;
    if (e >= ET) return;
    int s, d;
    if (e < E) { s = ei[e]; d = ei[E + e]; } else { s = d = e - E; }
    float alpha = ebuf[e] / fmaxf(denom[d], 1e-16f);
    const float* hs = h + (size_t)s * F;
    float* ad = agg + (size_t)d * F;
#pragma unroll
    for (int j = 0; j < F / 32; ++j) {
        int f = lane + 32 * j;
        atomicAdd(ad + f, alpha * hs[f]);
    }
}

// ---------------------------------------------------------------------------
// Bias add (+ optional ReLU), in place
// ---------------------------------------------------------------------------
template <int F, bool RELU>
__global__ void bias_act(float* __restrict__ z, const float* __restrict__ b, int n) {
    int i = blockIdx.x * blockDim.x + threadIdx.x;
    if (i >= n * F) return;
    float v = z[i] + b[i & (F - 1)];
    if (RELU) v = fmaxf(v, 0.f);
    z[i] = v;
}

// ---------------------------------------------------------------------------
// Decode: out[l] = dot(z2[src_l], z2[dst_l]) over 32 features.
// One wave per candidate edge, butterfly reduction.
// ---------------------------------------------------------------------------
__global__ void decode_dot(const int* __restrict__ eli, int L,
                           const float* __restrict__ z, float* __restrict__ out) {
    int gt = blockIdx.x * blockDim.x + threadIdx.x;
    int l = gt >> 5;
    int lane = gt & 31;
    if (l >= L) return;
    int s = eli[l], d = eli[L + l];
    float p = z[(size_t)s * 32 + lane] * z[(size_t)d * 32 + lane];
#pragma unroll
    for (int off = 16; off; off >>= 1) p += __shfl_xor(p, off, 32);
    if (lane == 0) out[l] = p;
}

// ---------------------------------------------------------------------------
// Host-side launcher
// ---------------------------------------------------------------------------
static inline unsigned cdiv(long long a, long long b) { return (unsigned)((a + b - 1) / b); }

extern "C" void kernel_launch(void* const* d_in, const int* in_sizes, int n_in,
                              void* d_out, int out_size, void* d_ws, size_t ws_size,
                              hipStream_t stream) {
    const float* x    = (const float*)d_in[0];
    const int*   ei   = (const int*)d_in[1];
    const int*   eli  = (const int*)d_in[2];
    const float* W1   = (const float*)d_in[3];
    const float* a_s1 = (const float*)d_in[4];
    const float* a_d1 = (const float*)d_in[5];
    const float* b1   = (const float*)d_in[6];
    const float* W2   = (const float*)d_in[7];
    const float* a_s2 = (const float*)d_in[8];
    const float* a_d2 = (const float*)d_in[9];
    const float* b2   = (const float*)d_in[10];

    const int FMID = in_sizes[4];          // 64
    const int FIN  = in_sizes[3] / FMID;   // 128
    const int Nn   = in_sizes[0] / FIN;    // 50000
    const int E    = in_sizes[1] / 2;      // 800000
    const int L    = in_sizes[2] / 2;      // 262144
    const int ET   = E + Nn;

    // Workspace layout: f32 region first, then bf16 region
    float* ws    = (float*)d_ws;
    float* h1    = ws; ws += (size_t)Nn * 64;
    float* agg1  = ws; ws += (size_t)Nn * 64;   // becomes z1 in place
    float* h2    = ws; ws += (size_t)Nn * 32;
    float* agg2  = ws; ws += (size_t)Nn * 32;   // becomes z2 in place
    float* asv   = ws; ws += Nn;
    float* adv   = ws; ws += Nn;
    float* maxv  = ws; ws += Nn;
    float* denom = ws; ws += Nn;
    float* ebuf  = ws; ws += ET;
    __bf16* wb   = (__bf16*)ws;
    __bf16* xbf  = wb; wb += (size_t)Nn * FIN;  // bf16 copy of x
    __bf16* z1bf = wb; wb += (size_t)Nn * 64;   // bf16 copy of z1
    __bf16* wp1  = wb; wb += 128 * 64;          // packed W1 fragments
    __bf16* wp2  = wb; wb += 64 * 32;           // packed W2 fragments
    (void)ws_size; (void)n_in; (void)out_size;

    const float NEG_INF = -__builtin_huge_valf();
    const unsigned T = 256;

    // ---------------- Layer 1 (F = 64) ----------------
    fill_f32<<<cdiv((long long)Nn * 64, T), T, 0, stream>>>(agg1, 0.f, Nn * 64);
    fill_f32<<<cdiv(Nn, T), T, 0, stream>>>(maxv, NEG_INF, Nn);
    fill_f32<<<cdiv(Nn, T), T, 0, stream>>>(denom, 0.f, Nn);

    cvt_f32_bf16<<<cdiv((long long)Nn * FIN / 4, T), T, 0, stream>>>(x, xbf, Nn * FIN / 4);
    prepack_w<128, 64><<<cdiv(128 * 64, T), T, 0, stream>>>(W1, wp1);
    gemm_wmma_bf16<128, 64><<<cdiv(Nn, 64), 128, 0, stream>>>(xbf, wp1, h1, Nn);
    attdot<64><<<cdiv(Nn, T), T, 0, stream>>>(h1, a_s1, a_d1, asv, adv, Nn);

    edge_score_max<<<cdiv(ET, T), T, 0, stream>>>(ei, E, Nn, asv, adv, ebuf, maxv);
    edge_exp_sum<<<cdiv(ET, T), T, 0, stream>>>(ei, E, Nn, maxv, ebuf, denom);
    edge_aggregate<64><<<cdiv((long long)ET * 32, T), T, 0, stream>>>(
        ei, E, Nn, h1, ebuf, denom, agg1);
    bias_act<64, true><<<cdiv((long long)Nn * 64, T), T, 0, stream>>>(agg1, b1, Nn);

    // ---------------- Layer 2 (F = 32) ----------------
    fill_f32<<<cdiv((long long)Nn * 32, T), T, 0, stream>>>(agg2, 0.f, Nn * 32);
    fill_f32<<<cdiv(Nn, T), T, 0, stream>>>(maxv, NEG_INF, Nn);
    fill_f32<<<cdiv(Nn, T), T, 0, stream>>>(denom, 0.f, Nn);

    cvt_f32_bf16<<<cdiv((long long)Nn * 64 / 4, T), T, 0, stream>>>(agg1, z1bf, Nn * 64 / 4);
    prepack_w<64, 32><<<cdiv(64 * 32, T), T, 0, stream>>>(W2, wp2);
    gemm_wmma_bf16<64, 32><<<cdiv(Nn, 64), 128, 0, stream>>>(z1bf, wp2, h2, Nn);
    attdot<32><<<cdiv(Nn, T), T, 0, stream>>>(h2, a_s2, a_d2, asv, adv, Nn);

    edge_score_max<<<cdiv(ET, T), T, 0, stream>>>(ei, E, Nn, asv, adv, ebuf, maxv);
    edge_exp_sum<<<cdiv(ET, T), T, 0, stream>>>(ei, E, Nn, maxv, ebuf, denom);
    edge_aggregate<32><<<cdiv((long long)ET * 32, T), T, 0, stream>>>(
        ei, E, Nn, h2, ebuf, denom, agg2);
    bias_act<32, false><<<cdiv((long long)Nn * 32, T), T, 0, stream>>>(agg2, b2, Nn);

    // ---------------- Decode ----------------
    decode_dot<<<cdiv((long long)L * 32, T), T, 0, stream>>>(eli, L, agg2, (float*)d_out);
}
